// DGNN_88476326297998
// MI455X (gfx1250) — compile-verified, matrix operand order checked by reading, not run on hardware
//
#include <hip/hip_runtime.h>

// ---------------------------------------------------------------------------
// Problem constants (from the reference)
// ---------------------------------------------------------------------------
constexpr int B    = 2048;
constexpr int HIST = 20;
constexpr int FEAT = 128;
constexpr int HID  = 128;
constexpr int OUT  = 64;
constexpr int ROWS = B * HIST;           // 40960 one-hop rows

// ---------------------------------------------------------------------------
// CDNA5 WMMA types
// ---------------------------------------------------------------------------
typedef __attribute__((ext_vector_type(16))) __bf16 v16bf;
typedef __attribute__((ext_vector_type(8)))  float  v8f;

struct FragU { union { v16bf v; uint4 u[2]; }; };

__device__ __forceinline__ v8f wmma_bf16(v16bf a, v16bf b, v8f c) {
    // 8 args: (neg_a, A, neg_b, B, c_mod, C, reuse_a, reuse_b)
    return __builtin_amdgcn_wmma_f32_16x16x32_bf16(
        false, a, false, b, (short)0, c, false, false);
}

// f32 -> bf16 round-to-nearest-even
__device__ __forceinline__ unsigned short f2bf(float f) {
    unsigned int u = __float_as_uint(f);
    u += 0x7FFFu + ((u >> 16) & 1u);
    return (unsigned short)(u >> 16);
}

// Load a 16x32 bf16 A-fragment (or mirrored B-fragment row) from a row-major
// [*,128] bf16 matrix.  ISA 7.12.2 16-bit A layout:
//   lanes 0-15 (M=lane):   halves 0-7 = K c*32+0..7,  halves 8-15 = K c*32+16..23
//   lanes 16-31 (M=lane-16): same +8 K offset
__device__ __forceinline__ v16bf load_frag(const unsigned short* __restrict__ base,
                                           int row, int kChunk, int lane) {
    const int koff = (lane & 16) ? 8 : 0;
    const unsigned short* p = base + row * 128 + kChunk * 32 + koff;
    FragU f;
    f.u[0] = *reinterpret_cast<const uint4*>(p);
    f.u[1] = *reinterpret_cast<const uint4*>(p + 16);
    return f.v;
}

// B-fragment for  out = in @ W^T :  B[k,n] = W[n,k];  n = nTile*16 + lane%16
__device__ __forceinline__ v16bf load_fragB(const unsigned short* __restrict__ Wbf,
                                            int nTile, int kChunk, int lane) {
    return load_frag(Wbf, nTile * 16 + (lane & 15), kChunk, lane);
}

// ---------------------------------------------------------------------------
// K0: convert weights f32 -> bf16
// ---------------------------------------------------------------------------
__global__ void __launch_bounds__(256)
convert_weights(const float* __restrict__ W0, const float* __restrict__ W2,
                const float* __restrict__ W4, const float* __restrict__ W6,
                unsigned short* W0bf, unsigned short* W2bf,
                unsigned short* W4bf, unsigned short* W6bf) {
    int i = blockIdx.x * blockDim.x + threadIdx.x;
    if (i < HID * FEAT) { W0bf[i] = f2bf(W0[i]); W2bf[i] = f2bf(W2[i]); }
    if (i < OUT * HID)  { W4bf[i] = f2bf(W4[i]); W6bf[i] = f2bf(W6[i]); }
}

// ---------------------------------------------------------------------------
// K1: per-b wave: soft1[b,:], agg1[b,:] = sum_h soft1*one_hop  (bf16),
//     and bf16 conversion of self_feat row.
// ---------------------------------------------------------------------------
__global__ void __launch_bounds__(32)
soft1_agg1_kernel(const float* __restrict__ self_feat,
                  const float* __restrict__ one_hop,
                  const float* __restrict__ e_time,
                  const float* __restrict__ his_time,
                  const float* __restrict__ delta,
                  float* __restrict__ soft1,
                  unsigned short* __restrict__ agg1_bf,
                  unsigned short* __restrict__ self_bf) {
    const int b    = blockIdx.x;
    const int lane = threadIdx.x;
    const float d  = delta[0];
    const float et = e_time[b];

    float v[HIST];
    float mx = -1e30f;
    #pragma unroll
    for (int k = 0; k < HIST; ++k) {
        v[k] = -d * (et - his_time[b * HIST + k]);
        mx = fmaxf(mx, v[k]);
    }
    float s = 0.f;
    #pragma unroll
    for (int k = 0; k < HIST; ++k) { v[k] = __expf(v[k] - mx); s += v[k]; }
    const float inv = 1.f / s;
    if (lane < HIST) soft1[b * HIST + lane] = v[lane] * inv;

    float ax = 0.f, ay = 0.f, az = 0.f, aw = 0.f;
    #pragma unroll
    for (int k = 0; k < HIST; ++k) {
        const float w = v[k] * inv;
        const float4 x = *reinterpret_cast<const float4*>(
            one_hop + (size_t)(b * HIST + k) * FEAT + lane * 4);
        ax += w * x.x; ay += w * x.y; az += w * x.z; aw += w * x.w;
    }
    ushort4 o; o.x = f2bf(ax); o.y = f2bf(ay); o.z = f2bf(az); o.w = f2bf(aw);
    *reinterpret_cast<ushort4*>(agg1_bf + (size_t)b * FEAT + lane * 4) = o;

    const float4 sf = *reinterpret_cast<const float4*>(
        self_feat + (size_t)b * FEAT + lane * 4);
    ushort4 so; so.x = f2bf(sf.x); so.y = f2bf(sf.y); so.z = f2bf(sf.z); so.w = f2bf(sf.w);
    *reinterpret_cast<ushort4*>(self_bf + (size_t)b * FEAT + lane * 4) = so;
}

// ---------------------------------------------------------------------------
// K2: per-(b,h) wave: soft2 softmax + weighted 2-hop aggregation (419 MB
// streaming pass -> memory roofline) + bf16 conversion of one_hop row.
// ---------------------------------------------------------------------------
__global__ void __launch_bounds__(256)
agg2_kernel(const float* __restrict__ one_hop,
            const float* __restrict__ two_hop,
            const float* __restrict__ his_time,
            const float* __restrict__ his_his_time,
            const float* __restrict__ delta,
            unsigned short* __restrict__ onehop_bf,
            unsigned short* __restrict__ agg2_bf) {
    const int wid  = blockIdx.x * (blockDim.x >> 5) + (threadIdx.x >> 5);
    const int lane = threadIdx.x & 31;
    if (wid >= ROWS) return;

    const float d  = delta[0];
    const float ht = his_time[wid];          // his_time[b,h] flat == wid
    const size_t base = (size_t)wid * HIST;  // (b*H+h)*H

    float v[HIST];
    float mx = -1e30f;
    #pragma unroll
    for (int k = 0; k < HIST; ++k) {
        v[k] = -d * (ht - his_his_time[base + k]);
        mx = fmaxf(mx, v[k]);
    }
    float s = 0.f;
    #pragma unroll
    for (int k = 0; k < HIST; ++k) { v[k] = __expf(v[k] - mx); s += v[k]; }
    const float inv = 1.f / s;

    float ax = 0.f, ay = 0.f, az = 0.f, aw = 0.f;
    #pragma unroll
    for (int k = 0; k < HIST; ++k) {
        if (k + 2 < HIST)
            __builtin_prefetch(two_hop + (base + k + 2) * FEAT + lane * 4, 0, 1);
        const float4 x = *reinterpret_cast<const float4*>(
            two_hop + (base + k) * FEAT + lane * 4);
        const float w = v[k] * inv;
        ax += w * x.x; ay += w * x.y; az += w * x.z; aw += w * x.w;
    }
    ushort4 o; o.x = f2bf(ax); o.y = f2bf(ay); o.z = f2bf(az); o.w = f2bf(aw);
    *reinterpret_cast<ushort4*>(agg2_bf + (size_t)wid * FEAT + lane * 4) = o;

    const float4 oh = *reinterpret_cast<const float4*>(
        one_hop + (size_t)wid * FEAT + lane * 4);
    ushort4 ob; ob.x = f2bf(oh.x); ob.y = f2bf(oh.y); ob.z = f2bf(oh.z); ob.w = f2bf(oh.w);
    *reinterpret_cast<ushort4*>(onehop_bf + (size_t)wid * FEAT + lane * 4) = ob;
}

// ---------------------------------------------------------------------------
// K3: fused layer-1 for all (b,h) rows via WMMA:
//   x_one_s = relu(one_hop@W0^T + agg2@W2^T + b0 + b2)   [16x128 tile/wave]
//   acc[b,:] = sum_h soft1[b,h] * x_one_s[b,h,:]          (in-register reduce)
// 2 waves/block, block owns one b (20 rows, tile2 padded with zero weights).
// ---------------------------------------------------------------------------
__global__ void __launch_bounds__(64)
layer1_onehop_kernel(const unsigned short* __restrict__ onehop_bf,
                     const unsigned short* __restrict__ agg2_bf,
                     const unsigned short* __restrict__ W0bf,
                     const unsigned short* __restrict__ W2bf,
                     const float* __restrict__ b0,
                     const float* __restrict__ b2,
                     const float* __restrict__ soft1,
                     unsigned short* __restrict__ acc_bf) {
    __shared__ float red[2][HID];
    const int b     = blockIdx.x;
    const int wave  = threadIdx.x >> 5;
    const int lane  = threadIdx.x & 31;
    const int mBase = (lane & 16) ? 8 : 0;
    const int m     = lane & 15;

    int row = b * HIST + wave * 16 + m;
    if (row > ROWS - 1) row = ROWS - 1;    // clamp OOB pad rows (weight = 0)

    v16bf A0[4], A1[4];
    #pragma unroll
    for (int c = 0; c < 4; ++c) {
        A0[c] = load_frag(onehop_bf, row, c, lane);
        A1[c] = load_frag(agg2_bf,  row, c, lane);
    }

    v8f C[8] = {};
    #pragma unroll
    for (int n = 0; n < 8; ++n) {
        #pragma unroll
        for (int c = 0; c < 4; ++c) {
            C[n] = wmma_bf16(A0[c], load_fragB(W0bf, n, c, lane), C[n]);
            C[n] = wmma_bf16(A1[c], load_fragB(W2bf, n, c, lane), C[n]);
        }
    }

    // per-lane M rows covered by D fragments: mBase + r  (r = 0..7)
    float w[8];
    #pragma unroll
    for (int r = 0; r < 8; ++r) {
        const int h = wave * 16 + mBase + r;
        w[r] = (h < HIST) ? soft1[b * HIST + h] : 0.f;
    }

    #pragma unroll
    for (int n = 0; n < 8; ++n) {
        const int N = n * 16 + m;
        const float bias = b0[N] + b2[N];
        float s = 0.f;
        #pragma unroll
        for (int r = 0; r < 8; ++r) {
            float x = C[n][r] + bias;
            x = x > 0.f ? x : 0.f;           // relu
            s += w[r] * x;
        }
        s += __shfl_xor(s, 16, 32);          // combine M 0-7 with M 8-15
        if (lane < 16) red[wave][N] = s;
    }
    __syncthreads();
    for (int i = threadIdx.x; i < HID; i += 64)
        acc_bf[(size_t)b * HID + i] = f2bf(red[0][i] + red[1][i]);
}

// ---------------------------------------------------------------------------
// K4: fused self path + layer 2 (one 16-row tile of B per wave):
//   x_s_one = relu(self@W0^T + agg1@W2^T + b0 + b2)   -> LDS (bf16)
//   out     = x_s_one@W4^T + acc@W6^T + b4 + b6       [16x64]
// ---------------------------------------------------------------------------
__global__ void __launch_bounds__(32)
layer2_kernel(const unsigned short* __restrict__ self_bf,
              const unsigned short* __restrict__ agg1_bf,
              const unsigned short* __restrict__ acc_bf,
              const unsigned short* __restrict__ W0bf,
              const unsigned short* __restrict__ W2bf,
              const unsigned short* __restrict__ W4bf,
              const unsigned short* __restrict__ W6bf,
              const float* __restrict__ b0, const float* __restrict__ b2,
              const float* __restrict__ b4, const float* __restrict__ b6,
              float* __restrict__ out) {
    __shared__ unsigned short xso[16 * HID];   // 4 KB bf16 restage
    const int tile  = blockIdx.x;              // 0..127
    const int lane  = threadIdx.x;
    const int m     = lane & 15;
    const int mBase = (lane & 16) ? 8 : 0;
    const int rowBase = tile * 16;

    v16bf A0[4], A1[4];
    #pragma unroll
    for (int c = 0; c < 4; ++c) {
        A0[c] = load_frag(self_bf, rowBase + m, c, lane);
        A1[c] = load_frag(agg1_bf, rowBase + m, c, lane);
    }
    v8f C[8] = {};
    #pragma unroll
    for (int n = 0; n < 8; ++n) {
        #pragma unroll
        for (int c = 0; c < 4; ++c) {
            C[n] = wmma_bf16(A0[c], load_fragB(W0bf, n, c, lane), C[n]);
            C[n] = wmma_bf16(A1[c], load_fragB(W2bf, n, c, lane), C[n]);
        }
    }
    // relu + bias, restage as bf16 A-matrix in LDS
    #pragma unroll
    for (int n = 0; n < 8; ++n) {
        const int N = n * 16 + m;
        const float bias = b0[N] + b2[N];
        #pragma unroll
        for (int r = 0; r < 8; ++r) {
            float x = C[n][r] + bias;
            x = x > 0.f ? x : 0.f;
            xso[(mBase + r) * HID + N] = f2bf(x);
        }
    }
    __syncthreads();

    v16bf A2[4], A3[4];
    #pragma unroll
    for (int c = 0; c < 4; ++c) {
        A2[c] = load_frag(xso, m, c, lane);                 // LDS (ds_load_b128)
        A3[c] = load_frag(acc_bf, rowBase + m, c, lane);
    }
    v8f D[4] = {};
    #pragma unroll
    for (int n = 0; n < 4; ++n) {
        #pragma unroll
        for (int c = 0; c < 4; ++c) {
            D[n] = wmma_bf16(A2[c], load_fragB(W4bf, n, c, lane), D[n]);
            D[n] = wmma_bf16(A3[c], load_fragB(W6bf, n, c, lane), D[n]);
        }
    }
    #pragma unroll
    for (int n = 0; n < 4; ++n) {
        const int N = n * 16 + m;
        const float bias = b4[N] + b6[N];
        #pragma unroll
        for (int r = 0; r < 8; ++r)
            out[(size_t)(rowBase + mBase + r) * OUT + N] = D[n][r] + bias;
    }
}

// ---------------------------------------------------------------------------
// Workspace layout (bytes, all 16B aligned)
// ---------------------------------------------------------------------------
constexpr size_t OFF_SOFT1  = 0;                                   // B*H f32
constexpr size_t OFF_SELFBF = OFF_SOFT1  + (size_t)B * HIST * 4;   // B*128 bf16
constexpr size_t OFF_AGG1BF = OFF_SELFBF + (size_t)B * FEAT * 2;
constexpr size_t OFF_OHBF   = OFF_AGG1BF + (size_t)B * FEAT * 2;   // ROWS*128 bf16
constexpr size_t OFF_AGG2BF = OFF_OHBF   + (size_t)ROWS * FEAT * 2;
constexpr size_t OFF_ACCBF  = OFF_AGG2BF + (size_t)ROWS * FEAT * 2;
constexpr size_t OFF_W0BF   = OFF_ACCBF  + (size_t)B * HID * 2;
constexpr size_t OFF_W2BF   = OFF_W0BF   + (size_t)HID * FEAT * 2;
constexpr size_t OFF_W4BF   = OFF_W2BF   + (size_t)HID * FEAT * 2;
constexpr size_t OFF_W6BF   = OFF_W4BF   + (size_t)OUT * HID * 2;

extern "C" void kernel_launch(void* const* d_in, const int* in_sizes, int n_in,
                              void* d_out, int out_size, void* d_ws, size_t ws_size,
                              hipStream_t stream) {
    const float* self_feat = (const float*)d_in[0];
    const float* one_hop   = (const float*)d_in[1];
    const float* two_hop   = (const float*)d_in[2];
    const float* e_time    = (const float*)d_in[3];
    const float* his_time  = (const float*)d_in[4];
    const float* his_his   = (const float*)d_in[5];
    const float* W0 = (const float*)d_in[6];
    const float* b0 = (const float*)d_in[7];
    const float* W2 = (const float*)d_in[8];
    const float* b2 = (const float*)d_in[9];
    const float* W4 = (const float*)d_in[10];
    const float* b4 = (const float*)d_in[11];
    const float* W6 = (const float*)d_in[12];
    const float* b6 = (const float*)d_in[13];
    const float* delta = (const float*)d_in[14];
    float* out = (float*)d_out;

    char* ws = (char*)d_ws;
    float*          soft1   = (float*)(ws + OFF_SOFT1);
    unsigned short* self_bf = (unsigned short*)(ws + OFF_SELFBF);
    unsigned short* agg1_bf = (unsigned short*)(ws + OFF_AGG1BF);
    unsigned short* oh_bf   = (unsigned short*)(ws + OFF_OHBF);
    unsigned short* agg2_bf = (unsigned short*)(ws + OFF_AGG2BF);
    unsigned short* acc_bf  = (unsigned short*)(ws + OFF_ACCBF);
    unsigned short* W0bf    = (unsigned short*)(ws + OFF_W0BF);
    unsigned short* W2bf    = (unsigned short*)(ws + OFF_W2BF);
    unsigned short* W4bf    = (unsigned short*)(ws + OFF_W4BF);
    unsigned short* W6bf    = (unsigned short*)(ws + OFF_W6BF);

    convert_weights<<<64, 256, 0, stream>>>(W0, W2, W4, W6, W0bf, W2bf, W4bf, W6bf);
    soft1_agg1_kernel<<<B, 32, 0, stream>>>(self_feat, one_hop, e_time, his_time,
                                            delta, soft1, agg1_bf, self_bf);
    agg2_kernel<<<ROWS / 8, 256, 0, stream>>>(one_hop, two_hop, his_time, his_his,
                                              delta, oh_bf, agg2_bf);
    layer1_onehop_kernel<<<B, 64, 0, stream>>>(oh_bf, agg2_bf, W0bf, W2bf,
                                               b0, b2, soft1, acc_bf);
    layer2_kernel<<<B / 16, 32, 0, stream>>>(self_bf, agg1_bf, acc_bf,
                                             W0bf, W2bf, W4bf, W6bf,
                                             b0, b2, b4, b6, out);
}
